// MoUNets_46755013984328
// MI455X (gfx1250) — matrix operand order repaired and checked by reading, not compiled, not run on hardware
//
#include <hip/hip_runtime.h>
#include <hip/hip_bf16.h>

typedef __attribute__((ext_vector_type(16))) _Float16 v16h;
typedef __attribute__((ext_vector_type(8)))  _Float16 v8h;
typedef __attribute__((ext_vector_type(8)))  float    v8f;

#define BN_EPS 1e-3f

// ---------------------------------------------------------------------------
// Weight prep: [27][CI][CO] f32  ->  [27][CI/32][CO][32] f16
// B-fragment order within the 32-chunk is natural channel order (lane group
// 0-15 reads K=0..15, lanes 16-31 read K=16..31 => identity permutation).
// ---------------------------------------------------------------------------
template<int CI, int CO>
__global__ void prep_w_kernel(const float* __restrict__ k, _Float16* __restrict__ out) {
    constexpr int S = CI / 32;
    int tid = blockIdx.x * blockDim.x + threadIdx.x;
    if (tid >= 27 * CI * CO) return;
    int j    = tid & 31;
    int rest = tid >> 5;          // [27][S][CO]
    int co   = rest % CO; rest /= CO;
    int s    = rest % S;
    int tap  = rest / S;
    int ci   = s * 32 + j;
    out[tid] = (_Float16)k[((size_t)(tap * CI + ci)) * CO + co];
}

// ---------------------------------------------------------------------------
// Block 0: direct conv (Cin=1) + bias + BN + ReLU, f32 in -> f16 out
// out: [8,32,32,32,32] NDHWC
// ---------------------------------------------------------------------------
__global__ void conv0_kernel(const float* __restrict__ x, const float* __restrict__ kw,
                             const float* __restrict__ bias,
                             const float* __restrict__ g, const float* __restrict__ be,
                             const float* __restrict__ mn, const float* __restrict__ vr,
                             _Float16* __restrict__ out) {
    int tid = blockIdx.x * 256 + threadIdx.x;        // 8*32^3*32 = 8388608
    int co = tid & 31;
    int m  = tid >> 5;                               // (b,od,oh,ow)
    int ow = m & 31;
    int oh = (m >> 5) & 31;
    int od = (m >> 10) & 31;
    int b  = m >> 15;
    float acc = 0.f;
    for (int kd = 0; kd < 3; kd++) {
        int id = 2 * od + kd; if (id >= 64) continue;
        for (int kh = 0; kh < 3; kh++) {
            int ih = 2 * oh + kh; if (ih >= 64) continue;
            #pragma unroll
            for (int kwi = 0; kwi < 3; kwi++) {
                int iw = 2 * ow + kwi;
                if (iw < 64) {
                    float xv = x[(((size_t)b * 64 + id) * 64 + ih) * 64 + iw];
                    float wv = kw[((kd * 3 + kh) * 3 + kwi) * 32 + co];
                    acc = fmaf(xv, wv, acc);
                }
            }
        }
    }
    acc += bias[co];
    float sc = g[co] * rsqrtf(vr[co] + BN_EPS);
    float v  = fmaxf((acc - mn[co]) * sc + be[co], 0.f);
    out[tid] = (_Float16)v;
}

// ---------------------------------------------------------------------------
// Blocks 1-3: implicit-GEMM conv via v_wmma_f32_16x16x32_f16.
// Each wave computes a 16(M) x 64(N) strip: 4 accumulators share one A
// fragment per K-step (4x reduction in activation HBM traffic vs 16x16
// tiles).  A: activations f16 NDHWC; B: pre-swizzled weights
// [tap][kstep][CO][32] f16 (L2-resident).  Fused bias + BN + ReLU epilogue.
// ---------------------------------------------------------------------------
template<int CI, int CO, int DI, int DO, bool OUT_F32>
__global__ __launch_bounds__(128)
void conv_wmma_kernel(const _Float16* __restrict__ xin, const _Float16* __restrict__ wsw,
                      const float* __restrict__ bias,
                      const float* __restrict__ g, const float* __restrict__ be,
                      const float* __restrict__ mn, const float* __restrict__ vr,
                      void* __restrict__ outv) {
    constexpr int S  = CI / 32;        // K-steps per tap
    constexpr int NT = CO / 16;        // N tiles (16-wide)
    constexpr int NG = NT / 4;         // N groups (64-wide strips)
    int lane  = threadIdx.x & 31;
    int wave  = threadIdx.x >> 5;
    int tile  = blockIdx.x * 4 + wave;
    int ng    = tile % NG;
    int tm    = tile / NG;
    int row   = lane & 15;
    int khalf = lane >> 4;

    // M decomposition for A loads (m = ((b*DO+od)*DO+oh)*DO+ow)
    int m  = tm * 16 + row;
    int ow = m % DO; int t1 = m / DO;
    int oh = t1 % DO; int t2 = t1 / DO;
    int od = t2 % DO;
    int b  = t2 / DO;

    int col = row;                      // N column within tile (lanes 0-15 / 16-31)
    v8f acc[4] = {};

    for (int kd = 0; kd < 3; kd++) {
        int id = 2 * od + kd;
        for (int kh = 0; kh < 3; kh++) {
            int ih = 2 * oh + kh;
            #pragma unroll
            for (int kwi = 0; kwi < 3; kwi++) {
                int iw  = 2 * ow + kwi;
                bool ok = (id < DI) & (ih < DI) & (iw < DI);
                int tap = (kd * 3 + kh) * 3 + kwi;
                size_t inoff = ((((size_t)b * DI + (ok ? id : 0)) * DI + (ok ? ih : 0)) * DI
                                + (ok ? iw : 0)) * CI;
                #pragma unroll
                for (int s = 0; s < S; s++) {
                    // A fragment: contiguous K order (matches B): lane khalf holds
                    // K = khalf*16 + [0..15]
                    v8h c0 = {}, c1 = {};
                    if (ok) {
                        c0 = *(const v8h*)(xin + inoff + s * 32 + khalf * 16);
                        c1 = *(const v8h*)(xin + inoff + s * 32 + khalf * 16 + 8);
                    }
                    v16h a;
                    #pragma unroll
                    for (int i = 0; i < 8; i++) { a[i] = c0[i]; a[8 + i] = c1[i]; }
                    // 4 N-tiles reuse the same A fragment
                    const _Float16* wb =
                        wsw + (((size_t)tap * S + s) * CO + ng * 64 + col) * 32 + khalf * 16;
                    #pragma unroll
                    for (int j = 0; j < 4; j++) {
                        v16h bf = *(const v16h*)(wb + (size_t)j * 16 * 32);
                        acc[j] = __builtin_amdgcn_wmma_f32_16x16x32_f16(
                            false, a, false, bf, (short)0, acc[j], false, false);
                    }
                }
            }
        }
    }

    // Epilogue: bias + BN + ReLU.  D layout: VGPR r of lane -> (M = r + 8*khalf, N = col)
    #pragma unroll
    for (int j = 0; j < 4; j++) {
        int co = (ng * 4 + j) * 16 + col;
        float sc = g[co] * rsqrtf(vr[co] + BN_EPS);
        float bs = bias[co], mm = mn[co], bb = be[co];
        #pragma unroll
        for (int r = 0; r < 8; r++) {
            int mo = tm * 16 + r + 8 * khalf;
            float v = fmaxf((acc[j][r] + bs - mm) * sc + bb, 0.f);
            size_t o = (size_t)mo * CO + co;
            if (OUT_F32) ((float*)outv)[o] = v;
            else         ((_Float16*)outv)[o] = (_Float16)v;
        }
    }
}

// ---------------------------------------------------------------------------
// Global average pool: h3 [8,4,4,4,256] f32 -> pooled [8,256]
// ---------------------------------------------------------------------------
__global__ void pool_kernel(const float* __restrict__ h3, float* __restrict__ pooled) {
    int tid = blockIdx.x * 256 + threadIdx.x;    // 2048
    if (tid >= 2048) return;
    int c = tid & 255;
    int b = tid >> 8;
    float s = 0.f;
    #pragma unroll 4
    for (int p = 0; p < 64; p++) s += h3[((size_t)b * 64 + p) * 256 + c];
    pooled[tid] = s * (1.f / 64.f);
}

// ---------------------------------------------------------------------------
// Gating head: 256->256 (BN,ReLU) -> 256->8 logits -> top-2 mask -> softmax.
// Writes gate weights [E=8][B=8].
// ---------------------------------------------------------------------------
__global__ void head_kernel(const float* __restrict__ pooled,
                            const float* __restrict__ w1k, const float* __restrict__ w1b,
                            const float* __restrict__ g, const float* __restrict__ be,
                            const float* __restrict__ mn, const float* __restrict__ vr,
                            const float* __restrict__ w2k, const float* __restrict__ w2b,
                            float* __restrict__ wgt) {
    __shared__ float sp[8 * 256];
    __shared__ float sh[256];
    __shared__ float slog[8][8];
    int t = threadIdx.x;   // 256
    for (int i = t; i < 2048; i += 256) sp[i] = pooled[i];
    __syncthreads();
    float sc = g[t] * rsqrtf(vr[t] + BN_EPS);
    for (int b = 0; b < 8; b++) {
        float acc = w1b[t];
        #pragma unroll 8
        for (int c = 0; c < 256; c++) acc = fmaf(sp[b * 256 + c], w1k[c * 256 + t], acc);
        sh[t] = fmaxf((acc - mn[t]) * sc + be[t], 0.f);
        __syncthreads();
        if (t < 8) {
            float a2 = w2b[t];
            for (int c = 0; c < 256; c++) a2 = fmaf(sh[c], w2k[c * 8 + t], a2);
            slog[b][t] = a2;
        }
        __syncthreads();
    }
    if (t < 8) {   // t = batch index
        float l[8];
        #pragma unroll
        for (int e = 0; e < 8; e++) l[e] = slog[t][e];
        float m1 = -1e30f, m2 = -1e30f;
        #pragma unroll
        for (int e = 0; e < 8; e++) {
            if (l[e] > m1) { m2 = m1; m1 = l[e]; }
            else if (l[e] > m2) { m2 = l[e]; }
        }
        float w[8]; float s = 0.f;
        #pragma unroll
        for (int e = 0; e < 8; e++) {
            if (l[e] >= m2) { w[e] = __expf(l[e] - m1); s += w[e]; }
            else w[e] = 0.f;
        }
        float inv = 1.f / s;
        #pragma unroll
        for (int e = 0; e < 8; e++) wgt[e * 8 + t] = w[e] * inv;
    }
}

// ---------------------------------------------------------------------------
// Weighted expert sum. blockIdx.y = batch -> weight test is wave-uniform,
// so zero-weight experts (6 of 8 after top-2) are skipped via execz branch:
// cuts expert-read HBM traffic ~4x.  out[b] = sum_e w[e][b]*expert[e][b].
// ---------------------------------------------------------------------------
__global__ void mix_kernel(const float* __restrict__ expert, const float* __restrict__ wgt,
                           float* __restrict__ out) {
    int b   = blockIdx.y;                                   // 0..7
    int idx = blockIdx.x * 256 + threadIdx.x;               // float4 index, 65536 per batch
    float4 acc = make_float4(0.f, 0.f, 0.f, 0.f);
    #pragma unroll
    for (int e = 0; e < 8; e++) {
        float we = wgt[e * 8 + b];                          // uniform per block
        if (we != 0.f) {
            const float4* p = (const float4*)(expert + ((size_t)e * 8 + b) * 262144) + idx;
            __builtin_prefetch(p + 256, 0, 1);              // global_prefetch_b8
            float4 v = *p;
            acc.x = fmaf(we, v.x, acc.x);
            acc.y = fmaf(we, v.y, acc.y);
            acc.z = fmaf(we, v.z, acc.z);
            acc.w = fmaf(we, v.w, acc.w);
        }
    }
    ((float4*)(out + (size_t)b * 262144))[idx] = acc;
}

// ---------------------------------------------------------------------------
// Launch
// ---------------------------------------------------------------------------
extern "C" void kernel_launch(void* const* d_in, const int* in_sizes, int n_in,
                              void* d_out, int out_size, void* d_ws, size_t ws_size,
                              hipStream_t stream) {
    (void)in_sizes; (void)n_in; (void)out_size; (void)ws_size;

    const float* x  = (const float*)d_in[0];
    const float* eo = (const float*)d_in[1];
    // per-block params: base = 2 + 6*i : k, b, bn_g, bn_b, bn_m, bn_v
    const float* ek[4]; const float* ebv[4]; const float* bg[4]; const float* bb[4];
    const float* bm[4]; const float* bv[4];
    for (int i = 0; i < 4; i++) {
        int base = 2 + 6 * i;
        ek[i]  = (const float*)d_in[base + 0];
        ebv[i] = (const float*)d_in[base + 1];
        bg[i]  = (const float*)d_in[base + 2];
        bb[i]  = (const float*)d_in[base + 3];
        bm[i]  = (const float*)d_in[base + 4];
        bv[i]  = (const float*)d_in[base + 5];
    }
    const float* w1k  = (const float*)d_in[26];
    const float* w1b  = (const float*)d_in[27];
    const float* wbng = (const float*)d_in[28];
    const float* wbnb = (const float*)d_in[29];
    const float* wbnm = (const float*)d_in[30];
    const float* wbnv = (const float*)d_in[31];
    const float* w2k  = (const float*)d_in[32];
    const float* w2b  = (const float*)d_in[33];
    float* out = (float*)d_out;

    // workspace layout (256B aligned offsets)
    char* ws = (char*)d_ws;
    _Float16* H0  = (_Float16*)(ws + 0);          // [8,32^3,32] f16  16.78 MB
    _Float16* H1  = (_Float16*)(ws + 16777216);   // [8,16^3,64] f16   4.19 MB
    _Float16* H2  = (_Float16*)(ws + 20971520);   // [8,8^3,128] f16   1.05 MB
    float*    H3  = (float*)   (ws + 22020096);   // [8,4^3,256] f32   0.52 MB
    float*    PL  = (float*)   (ws + 22544384);   // [8,256]
    float*    WG  = (float*)   (ws + 22552576);   // [8][8] gate weights
    _Float16* W1  = (_Float16*)(ws + 22552832);   // [27][1][64][32]
    _Float16* W2  = (_Float16*)(ws + 22663424);   // [27][2][128][32]
    _Float16* W3  = (_Float16*)(ws + 23105792);   // [27][4][256][32]

    // weight pre-swizzle (f32 -> f16, [tap][ci][co] -> [tap][kstep][co][32])
    prep_w_kernel<32, 64>  <<<216,  256, 0, stream>>>(ek[1], W1);
    prep_w_kernel<64, 128> <<<864,  256, 0, stream>>>(ek[2], W2);
    prep_w_kernel<128, 256><<<3456, 256, 0, stream>>>(ek[3], W3);

    // block 0: direct conv, f16 output
    conv0_kernel<<<32768, 256, 0, stream>>>(x, ek[0], ebv[0], bg[0], bb[0], bm[0], bv[0], H0);

    // blocks 1-3: WMMA implicit GEMM, 16x64 strip per wave, 4 waves per block.
    // grid = (Mtiles * Ngroups) / 4
    conv_wmma_kernel<32, 64, 32, 16, false><<<512, 128, 0, stream>>>(
        H0, W1, ebv[1], bg[1], bb[1], bm[1], bv[1], (void*)H1);
    conv_wmma_kernel<64, 128, 16, 8, false><<<128, 128, 0, stream>>>(
        H1, W2, ebv[2], bg[2], bb[2], bm[2], bv[2], (void*)H2);
    conv_wmma_kernel<128, 256, 8, 4, true><<<32, 128, 0, stream>>>(
        H2, W3, ebv[3], bg[3], bb[3], bm[3], bv[3], (void*)H3);

    // pool -> head (top-2 softmax gate) -> weighted mix
    pool_kernel<<<8, 256, 0, stream>>>(H3, PL);
    head_kernel<<<1, 256, 0, stream>>>(PL, w1k, w1b, wbng, wbnb, wbnm, wbnv, w2k, w2b, WG);
    mix_kernel<<<dim3(256, 8), 256, 0, stream>>>(eo, WG, out);
}
